// ModifiedHiFiGANGenerator_84696755077450
// MI455X (gfx1250) — matrix operand order repaired and checked by reading, not compiled, not run on hardware
//
#include <hip/hip_runtime.h>
#include <math.h>

// ---------------------------------------------------------------------------
// MI455X (gfx1250, wave32) ModifiedHiFiGAN generator.
// Heavy convs: implicit-GEMM over V_WMMA_F32_16X16X32_F16, fp16 operands /
// fp32 accum. Weight tiles staged global->LDS with GLOBAL_LOAD_ASYNC_TO_LDS
// (ASYNCcnt); activation tiles staged with 16B vector loads (kernel taps are
// padded to 8/16 so im2col runs are contiguous and division-free).
// ---------------------------------------------------------------------------

typedef __attribute__((ext_vector_type(16))) _Float16 v16h;
typedef __attribute__((ext_vector_type(8)))  _Float16 v8h;
typedef __attribute__((ext_vector_type(8)))  float    v8f;
typedef int v4i __attribute__((vector_size(4 * sizeof(int))));

#if __has_builtin(__builtin_amdgcn_global_load_async_to_lds_b128)
#define HAS_ASYNC_LDS 1
#else
#define HAS_ASYNC_LDS 0
#endif

__device__ __forceinline__ void wg_async_wait() {
#if HAS_ASYNC_LDS
#if __has_builtin(__builtin_amdgcn_s_wait_asynccnt)
  __builtin_amdgcn_s_wait_asynccnt(0);
#else
  asm volatile("s_wait_asynccnt 0x0" ::: "memory");
#endif
#endif
}

// ----------------------------- small utilities -----------------------------

__global__ void add_f32_kernel(const float* __restrict__ a,
                               const float* __restrict__ b,
                               float* __restrict__ o, int n) {
  int i = blockIdx.x * blockDim.x + threadIdx.x;
  if (i < n) o[i] = a[i] + b[i];
}

__global__ void f32_to_f16_kernel(const float* __restrict__ src,
                                  _Float16* __restrict__ dst, long n) {
  long i = blockIdx.x * (long)blockDim.x + threadIdx.x;
  if (i < n) dst[i] = (_Float16)src[i];
}

// pack fp32 [Co][Cin][ks] -> fp16 [Co][Cin][ksPad], zero-padded taps
__global__ void pack_w_tap_pad_kernel(const float* __restrict__ src,
                                      _Float16* __restrict__ dst,
                                      int Cin, int ks, int ksPad, long total) {
  long i = blockIdx.x * (long)blockDim.x + threadIdx.x;
  if (i >= total) return;
  int kt = (int)(i % ksPad); long t = i / ksPad;
  int ic = (int)(t % Cin);  long m = t / Cin;
  dst[i] = (kt < ks) ? (_Float16)src[(m * Cin + ic) * ks + kt] : (_Float16)0.f;
}

__global__ void copy_h_kernel(const _Float16* __restrict__ src,
                              _Float16* __restrict__ dst, int n) {
  int i = blockIdx.x * blockDim.x + threadIdx.x;
  if (i < n) dst[i] = src[i];
}

// out[b,j] = act( sum_e X[b,e]*W[j,e] + bias[j] );  mode: 0=id 1=sigmoid 2=softmax
__global__ __launch_bounds__(256) void matvec_act_kernel(
    const float* __restrict__ X, const float* __restrict__ W,
    const float* __restrict__ bias, float* __restrict__ out,
    int K, int J, int mode) {
  __shared__ float sh[512];
  __shared__ float red[2];
  int b = blockIdx.x;
  const float* x = X + (long)b * K;
  for (int j = threadIdx.x; j < J; j += blockDim.x) {
    const float* w = W + (long)j * K;
    float s = bias ? bias[j] : 0.f;
    for (int e = 0; e < K; ++e) s += w[e] * x[e];
    sh[j] = s;
  }
  __syncthreads();
  if (mode == 2) {
    if (threadIdx.x == 0) {
      float mx = -1e30f;
      for (int j = 0; j < J; ++j) mx = fmaxf(mx, sh[j]);
      float sum = 0.f;
      for (int j = 0; j < J; ++j) sum += __expf(sh[j] - mx);
      red[0] = mx; red[1] = sum;
    }
    __syncthreads();
    for (int j = threadIdx.x; j < J; j += blockDim.x)
      out[(long)b * J + j] = __expf(sh[j] - red[0]) / red[1];
  } else if (mode == 1) {
    for (int j = threadIdx.x; j < J; j += blockDim.x)
      out[(long)b * J + j] = 1.f / (1.f + __expf(-sh[j]));
  } else {
    for (int j = threadIdx.x; j < J; j += blockDim.x)
      out[(long)b * J + j] = sh[j];
  }
}

__global__ void pooled_mean_kernel(const _Float16* __restrict__ X,
                                   float* __restrict__ out, int total, int L) {
  int idx = blockIdx.x * blockDim.x + threadIdx.x;
  if (idx >= total) return;
  const _Float16* p = X + (long)idx * L;
  float s = 0.f;
  for (int t = 0; t < L; ++t) s += (float)p[t];
  out[idx] = s / (float)L;
}

__global__ void bias_sum_kernel(const float* __restrict__ bias4,
                                float* __restrict__ out, int Co) {
  for (int o = threadIdx.x; o < Co; o += blockDim.x) {
    float s = 0.f;
    for (int k = 0; k < 4; ++k) s += bias4[k * Co + o];
    out[o] = s;
  }
}

// Dynamic ODConv weight bank, O-padded rows and tap-padded kernel:
// Wd[b, o<OPad, i*ksPad+c]; zero for o>=O or c>=ks.
__global__ void odconv_w_kernel(const float* __restrict__ kern,  // [4,O,I,ks]
                                const float* __restrict__ ka,    // [B,4]
                                const float* __restrict__ sa,    // [B,ks]
                                const float* __restrict__ ia,    // [B,I]
                                const float* __restrict__ oa,    // [B,O]
                                _Float16* __restrict__ Wd,       // [B,OPad,I*ksPad]
                                int O, int OPad, int I, int ks, int ksPad,
                                long total) {
  long idx = blockIdx.x * (long)blockDim.x + threadIdx.x;
  if (idx >= total) return;
  int c = (int)(idx % ksPad); long t = idx / ksPad;
  int i = (int)(t % I); t /= I;
  int o = (int)(t % OPad); int b = (int)(t / OPad);
  if (o >= O || c >= ks) { Wd[idx] = (_Float16)0.f; return; }
  long ke = ((long)o * I + i) * ks + c;
  long stride = (long)O * I * ks;
  float s = 0.f;
  for (int k = 0; k < 4; ++k) s += kern[k * stride + ke] * ka[b * 4 + k];
  s *= sa[b * ks + c] * ia[b * I + i] * oa[b * O + o];
  Wd[idx] = (_Float16)s;
}

// ----------------------- WMMA implicit-GEMM conv1d -------------------------
// Block tile 64(M) x 64(N), K-chunk 32.  8 wave32 waves: 4(M) x 2(N-half),
// each wave holds two 16x16 accumulators (A fragment reused) -> 2 WMMA/iter.
// A tile: 1 async b128 global->LDS transfer per thread (padded weights, no
// bounds checks).  B tile: 1 contiguous 16B activation load per thread (KS is
// a multiple of 8 after tap padding; scalar fallback only at sequence edges).
template <int KS, int STRIDE, int PAD>
__global__ __launch_bounds__(256) void conv1d_wmma_kernel(
    const _Float16* __restrict__ Xh,   // [B, Cin, Lin]
    const _Float16* __restrict__ Wh,   // [(b)], rows x wrow halves, padded
    const float* __restrict__ bias,    // [Cout] or null
    const float* __restrict__ fsc,     // FiLM scale [B, Cout] or null
    const float* __restrict__ fsh,     // FiLM shift [B, Cout] or null
    _Float16* __restrict__ Yh,         // [B, Cout, Lout]
    int Cin, int Lin, int Cout, int Lout,
    int Ktot, int wrow, long wbs, int leaky) {
  static_assert(KS % 8 == 0, "tap-padded kernel size required");
  __shared__ _Float16 As[64 * 32];     // [m][k]
  __shared__ _Float16 Bs[64 * 32];     // [n][k]

  const int b     = blockIdx.z;
  const int mBase = blockIdx.y * 64;
  const int nBase = blockIdx.x * 64;
  const int tid   = threadIdx.x;
  const int lane  = tid & 31;
  const int wid   = tid >> 5;
  const int h     = (lane >> 4) & 1;
  const int l16   = lane & 15;
  const int wm    = wid >> 1;          // 0..3
  const int wn    = wid & 1;           // 0..1

  const _Float16* Wp = Wh + (long)b * wbs;
  const _Float16* Xb = Xh + (long)b * Cin * Lin;

  const int arow = tid >> 2, aseg = tid & 3;       // A: 64 rows x 4 b128 segs
  const int bn   = tid >> 2, brun = tid & 3;       // B: 64 rows x 4 runs of 8

  v8f acc0 = {}, acc1 = {};

  for (int k0 = 0; k0 < Ktot; k0 += 32) {
    // ---- A tile: one 16B global->LDS transfer per thread (ASYNCcnt path)
    {
      const _Float16* g = Wp + (long)(mBase + arow) * wrow + (k0 + aseg * 8);
      _Float16* l = &As[arow * 32 + aseg * 8];
#if HAS_ASYNC_LDS
      __builtin_amdgcn_global_load_async_to_lds_b128((v4i*)g, (v4i*)l, 0, 0);
#else
      v8h w; __builtin_memcpy(&w, g, 16);
      *(v8h*)l = w;
#endif
    }
    // ---- B tile (im2col): run of 8 halves stays inside one input channel
    {
      const int t_out = nBase + bn;
      const int gk0 = k0 + brun * 8;
      const int ic  = gk0 / KS;                 // KS power-of-two-ish: shifts
      const int kt0 = gk0 - ic * KS;            // 0 or 8
      const int xp0 = t_out * STRIDE - PAD + kt0;
      _Float16* l = &Bs[bn * 32 + brun * 8];
      if (t_out < Lout && xp0 >= 0 && xp0 + 7 < Lin) {
        v8h v; __builtin_memcpy(&v, Xb + (long)ic * Lin + xp0, 16);
        *(v8h*)l = v;
      } else {
#pragma unroll
        for (int e = 0; e < 8; ++e) {
          int xp = xp0 + e;
          _Float16 v = (_Float16)0.f;
          if (t_out < Lout && xp >= 0 && xp < Lin) v = Xb[(long)ic * Lin + xp];
          l[e] = v;
        }
      }
    }
    if (k0 + 32 < Ktot)  // speculative prefetch of next weight chunk
      __builtin_prefetch((const void*)(Wp + (long)(mBase + arow) * wrow + k0 + 32), 0, 1);
    wg_async_wait();
    __syncthreads();

    // ---- fragments in contiguous k order k = 16h + s, same for A and B (contiguous b128 runs in LDS)
    const _Float16* ar = &As[(wm * 16 + l16) * 32];
    v8h a0 = *(const v8h*)(ar + 16 * h);
    v8h a1 = *(const v8h*)(ar + 16 * h + 8);
    v16h af = __builtin_shufflevector(a0, a1, 0,1,2,3,4,5,6,7,8,9,10,11,12,13,14,15);

    const _Float16* br0 = &Bs[(wn * 32 + l16) * 32];
    const _Float16* br1 = &Bs[(wn * 32 + 16 + l16) * 32];
    v8h b00 = *(const v8h*)(br0 + 16 * h);
    v8h b01 = *(const v8h*)(br0 + 16 * h + 8);
    v8h b10 = *(const v8h*)(br1 + 16 * h);
    v8h b11 = *(const v8h*)(br1 + 16 * h + 8);
    v16h bf0 = __builtin_shufflevector(b00, b01, 0,1,2,3,4,5,6,7,8,9,10,11,12,13,14,15);
    v16h bf1 = __builtin_shufflevector(b10, b11, 0,1,2,3,4,5,6,7,8,9,10,11,12,13,14,15);

    acc0 = __builtin_amdgcn_wmma_f32_16x16x32_f16(false, af, false, bf0,
                                                  (short)0, acc0, false, false);
    acc1 = __builtin_amdgcn_wmma_f32_16x16x32_f16(false, af, false, bf1,
                                                  (short)0, acc1, false, false);
    __syncthreads();
  }

  // ---- epilogue: bias -> leaky -> FiLM, store fp16
  const int m0 = mBase + wm * 16 + 8 * h;
  const int n0 = nBase + wn * 32 + l16;
#pragma unroll
  for (int r = 0; r < 8; ++r) {
    int m = m0 + r;
    if (m < Cout) {
      float bv = bias ? bias[m] : 0.f;
      float sc = 1.f, sh = 0.f;
      if (fsc) { sc = fsc[b * Cout + m]; sh = fsh[b * Cout + m]; }
      float v0 = acc0[r] + bv;
      float v1 = acc1[r] + bv;
      if (leaky) {
        v0 = v0 > 0.f ? v0 : 0.1f * v0;
        v1 = v1 > 0.f ? v1 : 0.1f * v1;
      }
      v0 = sc * v0 + sh;
      v1 = sc * v1 + sh;
      long rowoff = ((long)b * Cout + m) * Lout;
      if (n0 < Lout)      Yh[rowoff + n0]      = (_Float16)v0;
      if (n0 + 16 < Lout) Yh[rowoff + n0 + 16] = (_Float16)v1;
    }
  }
}

// --------------------------- MRF GRC (C=32, L=32) --------------------------
__global__ __launch_bounds__(256) void grc_kernel(
    const _Float16* __restrict__ Xin,
    const float* __restrict__ conv_w,  // [32, 8, k]
    const float* __restrict__ conv_b,
    const float* __restrict__ loraA,   // [8, 8] (r, c)
    const float* __restrict__ loraB,   // [8, 8] (d, r)
    const float* __restrict__ alpha,   // [1]
    const float* __restrict__ mix_w,   // [32, 32]
    const float* __restrict__ mix_b,
    const float* __restrict__ gn_g, const float* __restrict__ gn_b,
    _Float16* __restrict__ Xout, int k, int d) {
  __shared__ float xs[32][32];
  __shared__ float pre[32][32];
  __shared__ float tA[4][8][32];
  __shared__ float ys[32][32];
  __shared__ float gstat[4][2];
  int b = blockIdx.x, tid = threadIdx.x;
  const _Float16* xb = Xin + b * 1024;
  for (int i = tid; i < 1024; i += 256) xs[i >> 5][i & 31] = (float)xb[i];
  __syncthreads();
  for (int i = tid; i < 4 * 8 * 32; i += 256) {
    int l = i & 31, r = (i >> 5) & 7, g = i >> 8;
    float s = 0.f;
    for (int c = 0; c < 8; ++c) s += loraA[r * 8 + c] * xs[g * 8 + c][l];
    tA[g][r][l] = s;
  }
  __syncthreads();
  int padw = ((k - 1) * d) >> 1;
  float al = alpha[0];
  for (int i = tid; i < 1024; i += 256) {
    int c = i >> 5, l = i & 31, g = c >> 3;
    float s = conv_b[c];
    for (int j = 0; j < 8; ++j)
      for (int kt = 0; kt < k; ++kt) {
        int xp = l - padw + kt * d;
        if (xp >= 0 && xp < 32) s += conv_w[(c * 8 + j) * k + kt] * xs[g * 8 + j][xp];
      }
    float lr = 0.f; int dc = c & 7;
    for (int r = 0; r < 8; ++r) lr += loraB[dc * 8 + r] * tA[g][r][l];
    pre[c][l] = s + al * lr;
  }
  __syncthreads();
  for (int i = tid; i < 1024; i += 256) {
    int c = i >> 5, l = i & 31;
    float s = mix_b[c];
    for (int cc = 0; cc < 32; ++cc) s += mix_w[c * 32 + cc] * pre[cc][l];
    ys[c][l] = s + xs[c][l];
  }
  __syncthreads();
  if (tid < 4) {
    float mu = 0.f, m2 = 0.f;
    for (int e = 0; e < 256; ++e) {
      float v = ys[tid * 8 + (e >> 5)][e & 31];
      mu += v; m2 += v * v;
    }
    mu *= (1.f / 256.f); m2 = m2 * (1.f / 256.f) - mu * mu;
    gstat[tid][0] = mu; gstat[tid][1] = rsqrtf(m2 + 1e-5f);
  }
  __syncthreads();
  for (int i = tid; i < 1024; i += 256) {
    int c = i >> 5, l = i & 31, g = c >> 3;
    float v = (ys[c][l] - gstat[g][0]) * gstat[g][1];
    v = v * gn_g[c] + gn_b[c];
    v = v > 0.f ? v : 0.1f * v;
    Xout[b * 1024 + i] = (_Float16)v;
  }
}

__global__ void res_film_kernel(const _Float16* __restrict__ cur,
                                const _Float16* __restrict__ res,
                                const float* __restrict__ sc,
                                const float* __restrict__ sh,
                                _Float16* __restrict__ out,
                                int C, int L, int total) {
  int idx = blockIdx.x * blockDim.x + threadIdx.x;
  if (idx >= total) return;
  int b = idx / (C * L), c = (idx / L) % C;
  float v = (float)cur[idx] + (float)res[idx];
  v = sc[b * C + c] * v + sh[b * C + c];
  out[idx] = (_Float16)v;
}

__global__ void out_conv_tanh_kernel(const _Float16* __restrict__ X,  // [B,32,32]
                                     const float* __restrict__ W,     // [1,32,7]
                                     const float* __restrict__ bias,
                                     float* __restrict__ out, int B) {
  int idx = blockIdx.x * blockDim.x + threadIdx.x;
  if (idx >= B * 32) return;
  int b = idx >> 5, t = idx & 31;
  float s = bias[0];
  for (int c = 0; c < 32; ++c)
    for (int kt = 0; kt < 7; ++kt) {
      int xp = t - 3 + kt;
      if (xp >= 0 && xp < 32) s += W[c * 7 + kt] * (float)X[(b * 32 + c) * 32 + xp];
    }
  out[idx] = tanhf(s);
}

// ------------------------------- orchestration -----------------------------
// d_in flattening assumption: top-level dict insertion order (mel, spk, emo,
// params); params flattened as JAX pytree (sorted dict keys, lists in order).

extern "C" void kernel_launch(void* const* d_in, const int* in_sizes, int n_in,
                              void* d_out, int out_size, void* d_ws, size_t ws_size,
                              hipStream_t stream) {
  (void)in_sizes; (void)n_in; (void)out_size; (void)ws_size;
  const int B = 16, T = 8192, EMBD = 448;
  const float* spk = (const float*)d_in[1];
  const float* emo = (const float*)d_in[2];
  const float* mel = (const float*)d_in[0];
  auto F = [&](int i) { return (const float*)d_in[i]; };

  const int FILM0 = 3;                 // per layer: sb, sw, tb, tw
  const int IDX_IN_B = 35, IDX_IN_W = 36;
  const int MRF0 = 37;                 // conv_b, conv_w, gn_b, gn_g, lora_A,
                                       // lora_B, lora_alpha, mix_b, mix_w
  const int IDX_OUT_B = 118, IDX_OUT_W = 119;
  const int UPS0 = 120;                // bias, ia_b, ia_w, ka_b, ka_w,
                                       // kernels, oa_b, oa_w, sa_b, sa_w

  char* base = (char*)d_ws; size_t off = 0;
  auto alloc = [&](size_t bytes) -> void* {
    void* p = base + off;
    off = (off + bytes + 255) & ~(size_t)255;
    return p;
  };

  const int FD[8] = {512, 256, 128, 64, 32, 32, 32, 32};
  float* emb = (float*)alloc((size_t)B * EMBD * 4);
  float* fsc[8]; float* fsh[8];
  for (int i = 0; i < 8; ++i) {
    fsc[i] = (float*)alloc((size_t)B * FD[i] * 4);
    fsh[i] = (float*)alloc((size_t)B * FD[i] * 4);
  }
  _Float16* melh = (_Float16*)alloc((size_t)B * 80 * T * 2);
  _Float16* winh = (_Float16*)alloc((size_t)512 * 80 * 8 * 2);  // tap-padded 7->8
  _Float16* x0h  = (_Float16*)alloc((size_t)B * 512 * 8192 * 2);
  _Float16* x1h  = (_Float16*)alloc((size_t)B * 256 * 1024 * 2);
  _Float16* x2h  = (_Float16*)alloc((size_t)B * 128 * 128 * 2);
  _Float16* x3h  = (_Float16*)alloc((size_t)B * 64 * 64 * 2);
  _Float16* x4h  = (_Float16*)alloc((size_t)B * 32 * 32 * 2);
  _Float16* wdyn = (_Float16*)alloc((size_t)B * 256 * 512 * 16 * 2);  // max stage
  float* pooled = (float*)alloc((size_t)B * 512 * 4);
  float* ka = (float*)alloc((size_t)B * 4 * 4);
  float* sa = (float*)alloc((size_t)B * 16 * 4);
  float* ia = (float*)alloc((size_t)B * 512 * 4);
  float* oa = (float*)alloc((size_t)B * 256 * 4);
  float* bsum = (float*)alloc(256 * 4);
  _Float16* mr0 = (_Float16*)alloc((size_t)B * 1024 * 2);
  _Float16* mr1 = (_Float16*)alloc((size_t)B * 1024 * 2);

  // 1) emb = spk + emo
  { int n = B * EMBD;
    add_f32_kernel<<<(n + 255) / 256, 256, 0, stream>>>(spk, emo, emb, n); }

  // 2) FiLM scale/shift
  for (int i = 0; i < 8; ++i) {
    matvec_act_kernel<<<B, 256, 0, stream>>>(emb, F(FILM0 + 4*i + 1), F(FILM0 + 4*i + 0),
                                             fsc[i], EMBD, FD[i], 0);
    matvec_act_kernel<<<B, 256, 0, stream>>>(emb, F(FILM0 + 4*i + 3), F(FILM0 + 4*i + 2),
                                             fsh[i], EMBD, FD[i], 0);
  }

  // 3) fp16 conversion / weight packing (taps 7->8, K-contiguous, zero pad)
  { long n = (long)B * 80 * T;
    f32_to_f16_kernel<<<(unsigned)((n + 255) / 256), 256, 0, stream>>>(mel, melh, n); }
  { long n = (long)512 * 80 * 8;
    pack_w_tap_pad_kernel<<<(unsigned)((n + 255) / 256), 256, 0, stream>>>(
        F(IDX_IN_W), winh, 80, 7, 8, n); }

  // 4) input conv (80->512, k7(pad8), pad3) + FiLM0, WMMA
  {
    dim3 g((T + 63) / 64, (512 + 63) / 64, B);
    conv1d_wmma_kernel<8, 1, 3><<<g, 256, 0, stream>>>(
        melh, winh, F(IDX_IN_B), fsc[0], fsh[0], x0h,
        80, T, 512, T, /*Ktot=*/640, /*wrow=*/640, /*wbs=*/0L, /*leaky=*/0);
  }

  // 5) ODConv stages (taps padded: 16->16, 4->8)
  const int CinS[4]   = {512, 256, 128, 64};
  const int CoutS[4]  = {256, 128, 64, 32};
  const int KsS[4]    = {16, 16, 4, 4};
  const int KsPadS[4] = {16, 16, 8, 8};
  const int LinS[4]   = {8192, 1024, 128, 64};
  const int LoutS[4]  = {1024, 128, 64, 32};
  _Float16* stageX[5] = {x0h, x1h, x2h, x3h, x4h};
  for (int s = 0; s < 4; ++s) {
    int Ci = CinS[s], Co = CoutS[s], ks = KsS[s], ksP = KsPadS[s];
    int Li = LinS[s], Lo = LoutS[s];
    int CoPad = (Co + 63) & ~63;
    int Ktot = Ci * ksP;                      // multiple of 32 for all stages
    int ub = UPS0 + 10 * s;
    { int n = B * Ci;
      pooled_mean_kernel<<<(n + 255) / 256, 256, 0, stream>>>(stageX[s], pooled, n, Li); }
    matvec_act_kernel<<<B, 256, 0, stream>>>(pooled, F(ub + 4), F(ub + 3), ka, Ci, 4, 2);
    matvec_act_kernel<<<B, 256, 0, stream>>>(pooled, F(ub + 9), F(ub + 8), sa, Ci, ks, 2);
    matvec_act_kernel<<<B, 256, 0, stream>>>(pooled, F(ub + 2), F(ub + 1), ia, Ci, Ci, 1);
    matvec_act_kernel<<<B, 256, 0, stream>>>(pooled, F(ub + 7), F(ub + 6), oa, Ci, Co, 1);
    bias_sum_kernel<<<1, 256, 0, stream>>>(F(ub + 0), bsum, Co);
    long tot = (long)B * CoPad * Ci * ksP;
    odconv_w_kernel<<<(unsigned)((tot + 255) / 256), 256, 0, stream>>>(
        F(ub + 5), ka, sa, ia, oa, wdyn, Co, CoPad, Ci, ks, ksP, tot);
    dim3 g((Lo + 63) / 64, (Co + 63) / 64, B);
    long wbs = (long)CoPad * Ktot;
    if (s < 2) {
      conv1d_wmma_kernel<16, 8, 4><<<g, 256, 0, stream>>>(
          stageX[s], wdyn, bsum, fsc[s + 1], fsh[s + 1], stageX[s + 1],
          Ci, Li, Co, Lo, Ktot, Ktot, wbs, /*leaky=*/1);
    } else {
      conv1d_wmma_kernel<8, 2, 1><<<g, 256, 0, stream>>>(
          stageX[s], wdyn, bsum, fsc[s + 1], fsh[s + 1], stageX[s + 1],
          Ci, Li, Co, Lo, Ktot, Ktot, wbs, /*leaky=*/1);
    }
  }

  // 6) MRF blocks (C=32, L=32)
  const int MK[3] = {3, 7, 11}, MD[3] = {1, 3, 5};
  _Float16* cur = x4h; _Float16* tmp = mr0; _Float16* res = mr1;
  int n1024 = B * 1024;
  for (int i = 0; i < 3; ++i) {
    copy_h_kernel<<<(n1024 + 255) / 256, 256, 0, stream>>>(cur, res, n1024);
    for (int j = 0; j < 3; ++j) {
      int pb = MRF0 + (i * 3 + j) * 9;
      grc_kernel<<<B, 256, 0, stream>>>(cur, F(pb + 1), F(pb + 0), F(pb + 4), F(pb + 5),
                                        F(pb + 6), F(pb + 8), F(pb + 7), F(pb + 3), F(pb + 2),
                                        tmp, MK[i], MD[j]);
      _Float16* t2 = cur; cur = tmp; tmp = t2;
    }
    res_film_kernel<<<(n1024 + 255) / 256, 256, 0, stream>>>(cur, res, fsc[5 + i], fsh[5 + i],
                                                             tmp, 32, 32, n1024);
    _Float16* t2 = cur; cur = tmp; tmp = t2;
  }

  // 7) output conv + tanh -> d_out (fp32 [16,1,32])
  out_conv_tanh_kernel<<<(B * 32 + 255) / 256, 256, 0, stream>>>(
      cur, F(IDX_OUT_W), F(IDX_OUT_B), (float*)d_out, B);
}
